// MoELayer_50405736186245
// MI455X (gfx1250) — matrix operand order repaired and checked by reading, not compiled, not run on hardware
//
#include <hip/hip_runtime.h>
#include <hip/hip_bf16.h>

#define D_MODEL 2048
#define D_FF    8192
#define N_EXP   4
#define TOKS    4096   // B*S = 2*2048

#define BM 32          // M tile (tokens)
#define BK 64          // K tile per stage
#define BN 256         // N tile (4 waves * 64 cols)

typedef __attribute__((ext_vector_type(16))) __bf16 v16bf;
typedef __attribute__((ext_vector_type(8)))  float  v8f;
typedef __attribute__((address_space(3))) unsigned short lds_u16;

union FragBF { v16bf v; uint4 q[2]; };

// fp32 -> bf16 round-to-nearest-even
__device__ __forceinline__ unsigned short f2bf(float f) {
    unsigned u = __float_as_uint(f);
    u = (u + 0x7FFFu + ((u >> 16) & 1u)) >> 16;
    return (unsigned short)u;
}
__device__ __forceinline__ unsigned pack2(float lo, float hi) {
    return (unsigned)f2bf(lo) | ((unsigned)f2bf(hi) << 16);
}

// A fragment: 16x32 bf16 window starting at element (row*BK + kwin)
__device__ __forceinline__ FragBF load_a(const unsigned short* As, int row, int kwin, int lane) {
    const int kh = (lane & 16) ? 8 : 0;
    FragBF a;
    a.q[0] = *(const uint4*)&As[row * BK + kwin + kh];
    a.q[1] = *(const uint4*)&As[row * BK + kwin + 16 + kh];
    return a;
}
// B fragment: 32x16 bf16, LDS stored [n][k], window at col*BK + kwin
__device__ __forceinline__ FragBF load_b(const unsigned short* Bs, int col, int kwin, int lane) {
    const int k0 = (lane & 16) ? 8 : 0;
    FragBF b;
    b.q[0] = *(const uint4*)&Bs[col * BK + kwin + k0];
    b.q[1] = *(const uint4*)&Bs[col * BK + kwin + k0 + 16];
    return b;
}

// ---------------------------------------------------------------------------
// Kernel 1: router. One wave32 per token.
// ---------------------------------------------------------------------------
__global__ __launch_bounds__(128) void moe_router(
    const float* __restrict__ x, const float* __restrict__ Wr,
    const float* __restrict__ br,
    int* __restrict__ counts, int* __restrict__ tok_list,
    float* __restrict__ topw)
{
    const int lane = threadIdx.x & 31;
    const int t    = blockIdx.x * 4 + (threadIdx.x >> 5);
    if (t >= TOKS) return;

    float a0 = 0.f, a1 = 0.f, a2 = 0.f, a3 = 0.f;
    const float* xr = x + (size_t)t * D_MODEL;
    for (int d = lane; d < D_MODEL; d += 32) {
        const float  xv = xr[d];
        const float4 w  = *(const float4*)&Wr[d * 4];
        a0 += xv * w.x; a1 += xv * w.y; a2 += xv * w.z; a3 += xv * w.w;
    }
    for (int off = 16; off; off >>= 1) {
        a0 += __shfl_xor(a0, off, 32);
        a1 += __shfl_xor(a1, off, 32);
        a2 += __shfl_xor(a2, off, 32);
        a3 += __shfl_xor(a3, off, 32);
    }
    if (lane == 0) {
        float l[4] = { a0 + br[0], a1 + br[1], a2 + br[2], a3 + br[3] };
        int   idx = 0; float lm = l[0];
        for (int e = 1; e < N_EXP; ++e) if (l[e] > lm) { lm = l[e]; idx = e; }
        float s = 0.f;
        for (int e = 0; e < N_EXP; ++e) s += __expf(l[e] - lm);
        const int pos = atomicAdd(&counts[idx], 1);
        tok_list[idx * TOKS + pos] = t;
        topw[t] = 1.0f / s;                 // exp(0)/sum
    }
}

// ---------------------------------------------------------------------------
// Kernel 2: gate+up. Block 256 thr = 8 waves as 2(M)x4(N); tile 32x256x64.
// B fragments software-pipelined so ds loads overlap WMMA issue.
// ---------------------------------------------------------------------------
__global__ __launch_bounds__(256) void moe_gemm1(
    const float* __restrict__ x, const float* __restrict__ Wg,
    const float* __restrict__ Wu,
    const int* __restrict__ counts, const int* __restrict__ tok_list,
    unsigned short* __restrict__ hbuf)
{
    const int e     = blockIdx.z;
    const int cnt   = counts[e];
    const int mbase = blockIdx.y * BM;
    if (mbase >= cnt) return;
    const int nbase = blockIdx.x * BN;

    __shared__ __align__(16) unsigned short As[BM * BK];   // 4 KB  [m][k]
    __shared__ __align__(16) unsigned short Bg[BN * BK];   // 32 KB [n][k]
    __shared__ __align__(16) unsigned short Bu[BN * BK];   // 32 KB
    __shared__ int tokIdx[BM];

    const int t    = threadIdx.x;
    const int lane = t & 31;
    const int wave = t >> 5;
    const int wm   = wave >> 2;                 // 0..1
    const int wn   = wave & 3;                  // 0..3

    if (t < BM) {
        const int row = mbase + t;
        tokIdx[t] = (row < cnt) ? tok_list[e * TOKS + row] : -1;
    }
    __syncthreads();

    // loop-invariant staging coordinates (branchless: pad rows read token 0)
    const int    am    = t >> 3;
    const int    ak    = (t & 7) * 8;
    const int    atok  = tokIdx[am];
    const size_t arow  = (size_t)(atok < 0 ? 0 : atok) * D_MODEL;
    const int    bn    = (t & 63) * 4;
    const int    bk0   = t >> 6;
    const float* wg_base = Wg + (size_t)e * D_MODEL * D_FF + nbase + bn;
    const float* wu_base = Wu + (size_t)e * D_MODEL * D_FF + nbase + bn;

    v8f accg[4] = {}, accu[4] = {};

    for (int kb = 0; kb < D_MODEL; kb += BK) {
        // ---- stage A: 8 fp32 -> one uint4 LDS store
        {
            const float4 x0 = *(const float4*)&x[arow + kb + ak];
            const float4 x1 = *(const float4*)&x[arow + kb + ak + 4];
            uint4 p;
            p.x = pack2(x0.x, x0.y); p.y = pack2(x0.z, x0.w);
            p.z = pack2(x1.x, x1.y); p.w = pack2(x1.z, x1.w);
            *(uint4*)&As[am * BK + ak] = p;
        }
        // ---- stage Bg/Bu: [k][n] fp32 -> [n][k] bf16
#pragma unroll 4
        for (int j = 0; j < 16; ++j) {
            const int    k    = bk0 + 4 * j;
            const size_t goff = (size_t)(kb + k) * D_FF;
            const float4 g4 = *(const float4*)&wg_base[goff];
            const float4 u4 = *(const float4*)&wu_base[goff];
            Bg[(bn + 0) * BK + k] = f2bf(g4.x);
            Bg[(bn + 1) * BK + k] = f2bf(g4.y);
            Bg[(bn + 2) * BK + k] = f2bf(g4.z);
            Bg[(bn + 3) * BK + k] = f2bf(g4.w);
            Bu[(bn + 0) * BK + k] = f2bf(u4.x);
            Bu[(bn + 1) * BK + k] = f2bf(u4.y);
            Bu[(bn + 2) * BK + k] = f2bf(u4.z);
            Bu[(bn + 3) * BK + k] = f2bf(u4.w);
        }
        if (kb + BK < D_MODEL) {               // global_prefetch next K tile
            __builtin_prefetch(&wg_base[(size_t)(kb + BK + bk0) * D_FF], 0, 1);
            __builtin_prefetch(&wu_base[(size_t)(kb + BK + bk0) * D_FF], 0, 1);
        }
        __syncthreads();

#pragma unroll
        for (int kw = 0; kw < BK; kw += 32) {
            const FragBF a = load_a(As, wm * 16 + (lane & 15), kw, lane);
            FragBF bg = load_b(Bg, wn * 64 + (lane & 15), kw, lane);
            FragBF bu = load_b(Bu, wn * 64 + (lane & 15), kw, lane);
#pragma unroll
            for (int sub = 0; sub < 4; ++sub) {
                FragBF bgn = bg, bun = bu;     // next-subtile prefetch
                if (sub < 3) {
                    const int coln = wn * 64 + (sub + 1) * 16 + (lane & 15);
                    bgn = load_b(Bg, coln, kw, lane);
                    bun = load_b(Bu, coln, kw, lane);
                }
                accg[sub] = __builtin_amdgcn_wmma_f32_16x16x32_bf16(
                    false, a.v, false, bg.v, (short)0, accg[sub], false, false);
                accu[sub] = __builtin_amdgcn_wmma_f32_16x16x32_bf16(
                    false, a.v, false, bu.v, (short)0, accu[sub], false, false);
                bg = bgn; bu = bun;
            }
        }
        __syncthreads();
    }

    // ---- epilogue: h = silu(g)*u -> bf16 scatter
    const int mrow0 = wm * 16 + ((lane & 16) ? 8 : 0);
    const int ncol  = lane & 15;
#pragma unroll
    for (int sub = 0; sub < 4; ++sub) {
        const int f = nbase + wn * 64 + sub * 16 + ncol;
#pragma unroll
        for (int r = 0; r < 8; ++r) {
            const int tok = tokIdx[mrow0 + r];
            if (tok >= 0) {
                const float g = accg[sub][r];
                const float u = accu[sub][r];
                const float h = (g / (1.0f + __expf(-g))) * u;
                hbuf[(size_t)tok * D_FF + f] = f2bf(h);
            }
        }
    }
}

// ---------------------------------------------------------------------------
// Kernel 3: down projection. A tile staged with GLOBAL_LOAD_ASYNC_TO_LDS_B128
// (ASYNCcnt path); B fragments software-pipelined.
// ---------------------------------------------------------------------------
__global__ __launch_bounds__(256) void moe_gemm2(
    const unsigned short* __restrict__ hbuf, const float* __restrict__ Wd,
    const int* __restrict__ counts, const int* __restrict__ tok_list,
    const float* __restrict__ topw, float* __restrict__ out)
{
    const int e     = blockIdx.z;
    const int cnt   = counts[e];
    const int mbase = blockIdx.y * BM;
    if (mbase >= cnt) return;
    const int nbase = blockIdx.x * BN;

    __shared__ __align__(16) unsigned short As[BM * BK];   // 4 KB
    __shared__ __align__(16) unsigned short Bd[BN * BK];   // 32 KB
    __shared__ int tokIdx[BM];

    const int t    = threadIdx.x;
    const int lane = t & 31;
    const int wave = t >> 5;
    const int wm   = wave >> 2;
    const int wn   = wave & 3;

    if (t < BM) {
        const int row = mbase + t;
        tokIdx[t] = (row < cnt) ? tok_list[e * TOKS + row] : -1;
    }
    __syncthreads();

    const int    am   = t >> 3;
    const int    ak   = (t & 7) * 8;
    const int    atok = tokIdx[am];
    const size_t arow = (size_t)(atok < 0 ? 0 : atok) * D_FF;
    const int    bn   = (t & 63) * 4;
    const int    bk0  = t >> 6;
    const float* wd_base = Wd + (size_t)e * D_FF * D_MODEL + nbase + bn;

    // loop-invariant LDS byte offset for this thread's async A-tile slot
    const unsigned lds_dst = (unsigned)(uintptr_t)(lds_u16*)&As[am * BK + ak];

    v8f acc[4] = {};

    for (int kb = 0; kb < D_FF; kb += BK) {
        // ---- stage A: async 16B global->LDS copy of bf16 h row (ASYNCcnt)
        {
            const unsigned long long gaddr =
                (unsigned long long)(const void*)&hbuf[arow + kb + ak];
            asm volatile("global_load_async_to_lds_b128 %0, %1, off"
                         :: "v"(lds_dst), "v"(gaddr) : "memory");
        }
        // ---- stage Bd
#pragma unroll 4
        for (int j = 0; j < 16; ++j) {
            const int    k    = bk0 + 4 * j;
            const float4 w4 = *(const float4*)&wd_base[(size_t)(kb + k) * D_MODEL];
            Bd[(bn + 0) * BK + k] = f2bf(w4.x);
            Bd[(bn + 1) * BK + k] = f2bf(w4.y);
            Bd[(bn + 2) * BK + k] = f2bf(w4.z);
            Bd[(bn + 3) * BK + k] = f2bf(w4.w);
        }
        if (kb + BK < D_FF)
            __builtin_prefetch(&wd_base[(size_t)(kb + BK + bk0) * D_MODEL], 0, 1);
        asm volatile("s_wait_asynccnt 0x0" ::: "memory");
        __syncthreads();

#pragma unroll
        for (int kw = 0; kw < BK; kw += 32) {
            const FragBF a = load_a(As, wm * 16 + (lane & 15), kw, lane);
            FragBF b = load_b(Bd, wn * 64 + (lane & 15), kw, lane);
#pragma unroll
            for (int sub = 0; sub < 4; ++sub) {
                FragBF bnx = b;
                if (sub < 3)
                    bnx = load_b(Bd, wn * 64 + (sub + 1) * 16 + (lane & 15), kw, lane);
                acc[sub] = __builtin_amdgcn_wmma_f32_16x16x32_bf16(
                    false, a.v, false, b.v, (short)0, acc[sub], false, false);
                b = bnx;
            }
        }
        __syncthreads();
    }

    // ---- epilogue: scale by routing weight, scatter fp32
    const int mrow0 = wm * 16 + ((lane & 16) ? 8 : 0);
    const int ncol  = lane & 15;
#pragma unroll
    for (int sub = 0; sub < 4; ++sub) {
        const int n = nbase + wn * 64 + sub * 16 + ncol;
#pragma unroll
        for (int r = 0; r < 8; ++r) {
            const int tok = tokIdx[mrow0 + r];
            if (tok >= 0)
                out[(size_t)tok * D_MODEL + n] = acc[sub][r] * topw[tok];
        }
    }
}

// ---------------------------------------------------------------------------
// ws layout = [counts 64B][tok_list 4*T ints][topw T floats][hbuf bf16 T*D_FF]
// ---------------------------------------------------------------------------
extern "C" void kernel_launch(void* const* d_in, const int* in_sizes, int n_in,
                              void* d_out, int out_size, void* d_ws, size_t ws_size,
                              hipStream_t stream) {
    const float* x  = (const float*)d_in[0];
    const float* Wr = (const float*)d_in[1];
    const float* br = (const float*)d_in[2];
    const float* Wg = (const float*)d_in[3];
    const float* Wu = (const float*)d_in[4];
    const float* Wd = (const float*)d_in[5];
    float* out = (float*)d_out;

    char* ws = (char*)d_ws;
    int*   counts   = (int*)ws;
    int*   tok_list = (int*)(ws + 64);
    float* topw     = (float*)(ws + 64 + (size_t)N_EXP * TOKS * sizeof(int));
    size_t hoff = 64 + (size_t)N_EXP * TOKS * sizeof(int) + (size_t)TOKS * sizeof(float);
    hoff = (hoff + 255) & ~(size_t)255;
    unsigned short* hbuf = (unsigned short*)(ws + hoff);

    hipMemsetAsync(counts, 0, 64, stream);

    moe_router<<<TOKS / 4, 128, 0, stream>>>(x, Wr, br, counts, tok_list, topw);

    dim3 g1(D_FF / BN, TOKS / BM, N_EXP);
    moe_gemm1<<<g1, 256, 0, stream>>>(x, Wg, Wu, counts, tok_list, hbuf);

    dim3 g2(D_MODEL / BN, TOKS / BM, N_EXP);
    moe_gemm2<<<g2, 256, 0, stream>>>(hbuf, Wd, counts, tok_list, topw, out);
}
